// IntraSetLayer_30640296689892
// MI455X (gfx1250) — matrix operand order repaired and checked by reading, not compiled, not run on hardware
//
#include <hip/hip_runtime.h>
#include <math.h>

// ---------------------------------------------------------------------------
// IntraSetLayer for MI455X (gfx1250, wave32, WMMA).
// GEMMs (F3, e, energy-bilinear, h) run on v_wmma_f32_16x16x32_f16.
// BatchNorm (training mode) handled with deterministic 2-phase reductions.
// ---------------------------------------------------------------------------

typedef __attribute__((ext_vector_type(16))) _Float16 v16h;
typedef __attribute__((ext_vector_type(8)))  float    v8f;

#define PPTS   4096
#define NPTS   32768                      // B*P
#define NKROW  (NPTS*16)                  // 524288

// ---- d_out layout (floats): outputs | trans | idx(as float) ----
#define TRANS_OFF (NPTS*16*128)           // 67108864
#define IDX_OFF   (TRANS_OFF + NKROW*2)   // 68157440

// ---- ws layout (floats), ~60 MB total required ----
#define BN1_SC     0                      // [2] scale, [2] shift
#define BN1_SH     2
#define BN2A_SC    4                      // [32] scale, [32] shift
#define BN2A_SH    36
#define BN2B_SC    68                     // [4] scale, [4] shift
#define BN2B_SH    72
#define PART1_OFF  128                    // 4096 blocks * 4
#define PART2A_OFF (PART1_OFF + 4096*4)   // 4096 * 64
#define PART2B_OFF (PART2A_OFF + 4096*64) // 4096 * 8
#define F3_OFF     (PART2B_OFF + 4096*8)  // 32768*128
#define ENERGY_OFF (F3_OFF + NPTS*128)    // 524288*16
#define H2_OFF     (ENERGY_OFF + NKROW*16)// 524288*4

__device__ __forceinline__ v8f wmma_f16(v16h a, v16h b, v8f c) {
  return __builtin_amdgcn_wmma_f32_16x16x32_f16(false, a, false, b, (short)0, c,
                                                false, false);
}

// ===========================================================================
// K0: F3 = features @ W3^T + b3   [32768x64] x [64x128], WMMA K=64
// ===========================================================================
__global__ __launch_bounds__(256) void k_f3(const float* __restrict__ features,
                                            const float* __restrict__ W3,
                                            const float* __restrict__ b3,
                                            float* __restrict__ ws) {
  const int wave = threadIdx.x >> 5, lane = threadIdx.x & 31;
  const int tile = blockIdx.x * 8 + wave;          // 2048 row tiles
  const int m = lane & 15, hi = lane >> 4;
  const int row = tile * 16 + m;

  v16h a0, a1;                                      // A: K=0..31 / 32..63
#pragma unroll
  for (int g = 0; g < 2; ++g)
#pragma unroll
    for (int j = 0; j < 8; ++j) {
      int k = g * 16 + hi * 8 + j;
      a0[g * 8 + j] = (_Float16)features[row * 64 + k];
      a1[g * 8 + j] = (_Float16)features[row * 64 + 32 + k];
    }
#pragma unroll
  for (int ct = 0; ct < 8; ++ct) {                  // 8 column tiles of 16
    const int ocol = ct * 16 + m;
    v16h b0, b1;
#pragma unroll
    for (int i = 0; i < 16; ++i) {
      int k = (i & 8) * 2 + hi * 8 + (i & 7);
      b0[i] = (_Float16)W3[ocol * 64 + k];
      b1[i] = (_Float16)W3[ocol * 64 + 32 + k];
    }
    v8f c = {};
    c = wmma_f16(a0, b0, c);
    c = wmma_f16(a1, b1, c);
    const float bias = b3[ocol];
#pragma unroll
    for (int r = 0; r < 8; ++r) {
      int mm = hi * 8 + r;
      ws[F3_OFF + (tile * 16 + mm) * 128 + ocol] = c[r] + bias;
    }
  }
}

// ===========================================================================
// K1: trans/idx outputs, BN1 partial stats, e = relu(...), energy via WMMA
// One wave per point (its 16 neighbors = one 16-row tile).
// ===========================================================================
__global__ __launch_bounds__(256) void k1_energy(
    const float* __restrict__ points, const float* __restrict__ features,
    const float* __restrict__ W1, const float* __restrict__ b1v,
    const float* __restrict__ Wb, const float* __restrict__ bbv,
    const float* __restrict__ Wp1, float* __restrict__ out,
    float* __restrict__ ws) {
  __shared__ _Float16 wb_lds[4096];   // Wb rearranged [i][j][o], f16
  __shared__ float tr_lds[8][16][2];
  __shared__ float e_lds[8][16][16];
  __shared__ float wsum[8][4];

  const int tid = threadIdx.x;
  for (int f = tid; f < 4096; f += 256) {           // Wb is [o][i][j]
    int o = f >> 8, i = (f >> 4) & 15, j = f & 15;
    wb_lds[i * 256 + j * 16 + o] = (_Float16)Wb[f];
  }
  __syncthreads();

  const int wave = tid >> 5, lane = tid & 31;
  const int n = blockIdx.x * 8 + wave;
  const int bi = n >> 12, pp = n & 4095;
  const int m = lane & 15, hi = lane >> 4;
  const int noff = (m < 8) ? (m - 8) : (m - 7);     // offsets [-8..-1,1..8]
  const int gidx = (bi << 12) + ((pp + noff + 4096) & 4095);
  const float t0 = points[gidx * 2 + 0] - points[n * 2 + 0];
  const float t1 = points[gidx * 2 + 1] - points[n * 2 + 1];
  const int row = n * 16 + m;
  if (hi == 0) {
    out[TRANS_OFF + row * 2 + 0] = t0;
    out[TRANS_OFF + row * 2 + 1] = t1;
    out[IDX_OFF + row] = (float)gidx;               // idx emitted as float
    tr_lds[wave][m][0] = t0;
    tr_lds[wave][m][1] = t1;
  }
  {  // deterministic BN1 partial sums (tp = trans @ Wp1^T)
    float tp0 = t0 * Wp1[0] + t1 * Wp1[1];
    float tp1 = t0 * Wp1[2] + t1 * Wp1[3];
    float r0 = hi ? 0.f : tp0, r1 = hi ? 0.f : tp1;
    float r2 = hi ? 0.f : tp0 * tp0, r3 = hi ? 0.f : tp1 * tp1;
#pragma unroll
    for (int d = 1; d < 32; d <<= 1) {
      r0 += __shfl_xor(r0, d); r1 += __shfl_xor(r1, d);
      r2 += __shfl_xor(r2, d); r3 += __shfl_xor(r3, d);
    }
    if (lane == 0) { wsum[wave][0] = r0; wsum[wave][1] = r1;
                     wsum[wave][2] = r2; wsum[wave][3] = r3; }
  }
  __syncthreads();
  if (tid < 4) {
    float s = 0.f;
#pragma unroll
    for (int w = 0; w < 8; ++w) s += wsum[w][tid];
    ws[PART1_OFF + blockIdx.x * 4 + tid] = s;
  }

  // ---- e = relu([trans | feat] @ W1^T + b1); feat part K=64 via WMMA ----
  v16h a0, a1;
#pragma unroll
  for (int g = 0; g < 2; ++g)
#pragma unroll
    for (int j = 0; j < 8; ++j) {
      int k = g * 16 + hi * 8 + j;
      a0[g * 8 + j] = (_Float16)features[gidx * 64 + k];
      a1[g * 8 + j] = (_Float16)features[gidx * 64 + 32 + k];
    }
  v16h fb0, fb1;                                    // B = W1[:,2:]^T
#pragma unroll
  for (int i = 0; i < 16; ++i) {
    int k = (i & 8) * 2 + hi * 8 + (i & 7);
    fb0[i] = (_Float16)W1[m * 66 + 2 + k];
    fb1[i] = (_Float16)W1[m * 66 + 2 + 32 + k];
  }
  v8f acc = {};
  acc = wmma_f16(a0, fb0, acc);
  acc = wmma_f16(a1, fb1, acc);
  const float w1t0 = W1[m * 66 + 0], w1t1 = W1[m * 66 + 1], b1o = b1v[m];
#pragma unroll
  for (int r = 0; r < 8; ++r) {
    int mm = hi * 8 + r;
    float e = acc[r] + tr_lds[wave][mm][0] * w1t0 +
              tr_lds[wave][mm][1] * w1t1 + b1o;
    e_lds[wave][mm][m] = fmaxf(e, 0.f);
  }
  __syncthreads();

  // ---- energy = (e ⊗ e) @ vec(Wb)^T : K = 256 = 8 x 32, exact ----
  float ev[16];
#pragma unroll
  for (int i = 0; i < 16; ++i) ev[i] = e_lds[wave][m][i];
  v8f en = {};
#pragma unroll
  for (int kf = 0; kf < 8; ++kf) {
    v16h aq, bq;
#pragma unroll
    for (int g = 0; g < 2; ++g)
#pragma unroll
      for (int j = 0; j < 8; ++j) {
        int kk = kf * 32 + g * 16 + hi * 8 + j;     // kk = i*16 + jj
        aq[g * 8 + j] = (_Float16)(ev[kk >> 4] * ev[kk & 15]);
      }
#pragma unroll
    for (int i = 0; i < 16; ++i) {
      int kk = kf * 32 + (i & 8) * 2 + hi * 8 + (i & 7);
      bq[i] = wb_lds[(kk >> 4) * 256 + (kk & 15) * 16 + m];
    }
    en = wmma_f16(aq, bq, en);
  }
  const float bbo = bbv[m];
#pragma unroll
  for (int r = 0; r < 8; ++r) {
    int mm = hi * 8 + r;
    ws[ENERGY_OFF + (n * 16 + mm) * 16 + m] = en[r] + bbo;
  }
}

// ===========================================================================
// Generic BN finalize: ordered (deterministic) reduction of block partials
// ===========================================================================
__global__ void k_bn_fin(const float* __restrict__ g, const float* __restrict__ b,
                         float* __restrict__ ws, int part_off, int nblk,
                         int sc_off, int C) {
  const int c = threadIdx.x;
  if (c >= C) return;
  float s = 0.f, q = 0.f;
  for (int bl = 0; bl < nblk; ++bl) {
    s += ws[part_off + bl * 2 * C + c];
    q += ws[part_off + bl * 2 * C + C + c];
  }
  const float inv = 1.0f / 524288.0f;
  const float mean = s * inv;
  const float var  = q * inv - mean * mean;         // biased, matches jnp.var
  const float sc = g[c] * rsqrtf(var + 1e-5f);
  ws[sc_off + c] = sc;
  ws[sc_off + C + c] = b[c] - mean * sc;
}

// ===========================================================================
// Shared tile: h = [energy | shrink] @ W2a^T  (K=32 exact, 2 WMMAs)
// ===========================================================================
__device__ __forceinline__ void compute_h_tile(
    int n, int lane, const float* __restrict__ ws, const float* __restrict__ out,
    const float* __restrict__ Wp1, const float* __restrict__ W2a,
    const float* foldA, const float* foldB, const float* foldC,
    v8f& c0, v8f& c1) {
  const int m = lane & 15, hi = lane >> 4;
  const int row = n * 16 + m;
  const float t0 = out[TRANS_OFF + row * 2 + 0];
  const float t1 = out[TRANS_OFF + row * 2 + 1];
  const float p0 = fmaxf((t0 * Wp1[0] + t1 * Wp1[1]) * ws[BN1_SC + 0] + ws[BN1_SH + 0], 0.f);
  const float p1 = fmaxf((t0 * Wp1[2] + t1 * Wp1[3]) * ws[BN1_SC + 1] + ws[BN1_SH + 1], 0.f);
  v16h a;
#pragma unroll
  for (int j = 0; j < 8; ++j) {
    const int kk = hi * 8 + j;
    a[j]     = (_Float16)ws[ENERGY_OFF + row * 16 + kk];          // K<16
    a[8 + j] = (_Float16)(p0 * foldA[kk] + p1 * foldB[kk] + foldC[kk]); // K>=16
  }
  v16h b0, b1;
#pragma unroll
  for (int i = 0; i < 16; ++i) {
    const int k = (i & 8) * 2 + hi * 8 + (i & 7);
    b0[i] = (_Float16)W2a[m * 32 + k];
    b1[i] = (_Float16)W2a[(16 + m) * 32 + k];
  }
  v8f z = {};
  c0 = wmma_f16(a, b0, z);
  c1 = wmma_f16(a, b1, z);
}

__device__ __forceinline__ void fold_wp2(int tid, const float* __restrict__ Wp2,
                                         const float* __restrict__ bp2,
                                         float* foldA, float* foldB, float* foldC) {
  if (tid < 16) {  // shrink[j] = sum_s p_embed[s*16+j] folds to 3 constants
    float A = 0.f, B = 0.f, C = 0.f;
#pragma unroll
    for (int s = 0; s < 8; ++s) {
      int c = s * 16 + tid;
      A += Wp2[c * 2]; B += Wp2[c * 2 + 1]; C += bp2[c];
    }
    foldA[tid] = A; foldB[tid] = B; foldC[tid] = C;
  }
}

// ===========================================================================
// K3: BN2a statistics of h (h not stored; recomputed later)
// ===========================================================================
__global__ __launch_bounds__(256) void k3_bn2a_stats(
    const float* __restrict__ Wp1, const float* __restrict__ Wp2,
    const float* __restrict__ bp2, const float* __restrict__ W2a,
    const float* __restrict__ out, float* __restrict__ ws) {
  __shared__ float foldA[16], foldB[16], foldC[16];
  __shared__ float pb[8][64];
  const int tid = threadIdx.x;
  fold_wp2(tid, Wp2, bp2, foldA, foldB, foldC);
  __syncthreads();
  const int wave = tid >> 5, lane = tid & 31;
  const int n = blockIdx.x * 8 + wave;
  const int m = lane & 15, hi = lane >> 4;
  v8f c0, c1;
  compute_h_tile(n, lane, ws, out, Wp1, W2a, foldA, foldB, foldC, c0, c1);
  float s0 = 0.f, q0 = 0.f, s1 = 0.f, q1 = 0.f;
#pragma unroll
  for (int r = 0; r < 8; ++r) {
    s0 += c0[r]; q0 += c0[r] * c0[r];
    s1 += c1[r]; q1 += c1[r] * c1[r];
  }
  s0 += __shfl_xor(s0, 16); q0 += __shfl_xor(q0, 16);
  s1 += __shfl_xor(s1, 16); q1 += __shfl_xor(q1, 16);
  if (hi == 0) {
    pb[wave][m] = s0; pb[wave][16 + m] = s1;
    pb[wave][32 + m] = q0; pb[wave][48 + m] = q1;
  }
  __syncthreads();
  if (tid < 64) {
    float s = 0.f;
#pragma unroll
    for (int w = 0; w < 8; ++w) s += pb[w][tid];
    ws[PART2A_OFF + blockIdx.x * 64 + tid] = s;
  }
}

// ===========================================================================
// K5: recompute h, apply BN2a+relu, h2 = g @ W2b^T (store) + BN2b stats
// ===========================================================================
__global__ __launch_bounds__(256) void k5_h2(
    const float* __restrict__ Wp1, const float* __restrict__ Wp2,
    const float* __restrict__ bp2, const float* __restrict__ W2a,
    const float* __restrict__ W2b, const float* __restrict__ out,
    float* __restrict__ ws) {
  __shared__ float foldA[16], foldB[16], foldC[16];
  __shared__ float g_lds[8][16][32];
  __shared__ float pw[8][8];
  const int tid = threadIdx.x;
  fold_wp2(tid, Wp2, bp2, foldA, foldB, foldC);
  __syncthreads();
  const int wave = tid >> 5, lane = tid & 31;
  const int n = blockIdx.x * 8 + wave;
  const int m = lane & 15, hi = lane >> 4;
  v8f c0, c1;
  compute_h_tile(n, lane, ws, out, Wp1, W2a, foldA, foldB, foldC, c0, c1);
  const float sA = ws[BN2A_SC + m],      hA = ws[BN2A_SH + m];
  const float sB = ws[BN2A_SC + 16 + m], hB = ws[BN2A_SH + 16 + m];
#pragma unroll
  for (int r = 0; r < 8; ++r) {
    const int mm = hi * 8 + r;
    g_lds[wave][mm][m]      = fmaxf(c0[r] * sA + hA, 0.f);
    g_lds[wave][mm][16 + m] = fmaxf(c1[r] * sB + hB, 0.f);
  }
  __syncthreads();
  float h2v[4] = {0.f, 0.f, 0.f, 0.f};
  if (hi == 0) {
#pragma unroll
    for (int c = 0; c < 4; ++c) {
      float a = 0.f;
#pragma unroll
      for (int cc = 0; cc < 32; ++cc) a += g_lds[wave][m][cc] * W2b[c * 32 + cc];
      h2v[c] = a;
      ws[H2_OFF + (n * 16 + m) * 4 + c] = a;
    }
  }
  float rr[8];
#pragma unroll
  for (int c = 0; c < 4; ++c) { rr[c] = h2v[c]; rr[4 + c] = h2v[c] * h2v[c]; }
#pragma unroll
  for (int d = 1; d < 32; d <<= 1)
#pragma unroll
    for (int c = 0; c < 8; ++c) rr[c] += __shfl_xor(rr[c], d);
  if (lane == 0) {
#pragma unroll
    for (int c = 0; c < 8; ++c) pw[wave][c] = rr[c];
  }
  __syncthreads();
  if (tid < 8) {
    float s = 0.f;
#pragma unroll
    for (int w = 0; w < 8; ++w) s += pw[w][tid];
    ws[PART2B_OFF + blockIdx.x * 8 + tid] = s;
  }
}

// ===========================================================================
// K7: logits -> in-lane softmax over 16 neighbors -> final outputs
// Half-wave per point; lane = output channel group j (Cout/share = 16).
// ===========================================================================
__global__ __launch_bounds__(256) void k7_out(
    const float* __restrict__ Wp1, const float* __restrict__ Wp2,
    const float* __restrict__ bp2, const float* __restrict__ W2c,
    const float* __restrict__ b2cv, float* __restrict__ out,
    const float* __restrict__ ws) {
  const int tid = threadIdx.x;
  const int wave = tid >> 5, lane = tid & 31;
  const int hi = lane >> 4, j = lane & 15;
  const int n = blockIdx.x * 16 + wave * 2 + hi;

  const float wc20 = W2c[j * 4 + 0], wc21 = W2c[j * 4 + 1];
  const float wc22 = W2c[j * 4 + 2], wc23 = W2c[j * 4 + 3];
  const float b2cj = b2cv[j];
  const float sb0 = ws[BN2B_SC + 0], sb1 = ws[BN2B_SC + 1];
  const float sb2 = ws[BN2B_SC + 2], sb3 = ws[BN2B_SC + 3];
  const float hb0 = ws[BN2B_SH + 0], hb1 = ws[BN2B_SH + 1];
  const float hb2 = ws[BN2B_SH + 2], hb3 = ws[BN2B_SH + 3];
  float wp2c0[8], wp2c1[8], wp2b[8];
#pragma unroll
  for (int s = 0; s < 8; ++s) {
    int c = s * 16 + j;
    wp2c0[s] = Wp2[c * 2]; wp2c1[s] = Wp2[c * 2 + 1]; wp2b[s] = bp2[c];
  }
  const float sc10 = ws[BN1_SC + 0], sh10 = ws[BN1_SH + 0];
  const float sc11 = ws[BN1_SC + 1], sh11 = ws[BN1_SH + 1];
  const float wp00 = Wp1[0], wp01 = Wp1[1], wp10 = Wp1[2], wp11 = Wp1[3];

  float lgt[16];
#pragma unroll
  for (int nb = 0; nb < 16; ++nb) {
    const int row = n * 16 + nb;
    const float g0 = fmaxf(ws[H2_OFF + row * 4 + 0] * sb0 + hb0, 0.f);
    const float g1 = fmaxf(ws[H2_OFF + row * 4 + 1] * sb1 + hb1, 0.f);
    const float g2 = fmaxf(ws[H2_OFF + row * 4 + 2] * sb2 + hb2, 0.f);
    const float g3 = fmaxf(ws[H2_OFF + row * 4 + 3] * sb3 + hb3, 0.f);
    lgt[nb] = g0 * wc20 + g1 * wc21 + g2 * wc22 + g3 * wc23 + b2cj;
  }
  float mx = lgt[0];
#pragma unroll
  for (int nb = 1; nb < 16; ++nb) mx = fmaxf(mx, lgt[nb]);
  float ssum = 0.f;
#pragma unroll
  for (int nb = 0; nb < 16; ++nb) { lgt[nb] = __expf(lgt[nb] - mx); ssum += lgt[nb]; }
  const float sinv = 1.f / ssum;

  const int bi = n >> 12, pp = n & 4095;
#pragma unroll
  for (int nb = 0; nb < 16; ++nb) {
    const int row = n * 16 + nb;
    const float wv = lgt[nb] * sinv;
    const float t0 = out[TRANS_OFF + row * 2 + 0];
    const float t1 = out[TRANS_OFF + row * 2 + 1];
    const float p0 = fmaxf((t0 * wp00 + t1 * wp01) * sc10 + sh10, 0.f);
    const float p1 = fmaxf((t0 * wp10 + t1 * wp11) * sc11 + sh11, 0.f);
    const int noff = (nb < 8) ? (nb - 8) : (nb - 7);
    const int gi = (bi << 12) + ((pp + noff + 4096) & 4095);
#pragma unroll
    for (int s = 0; s < 8; ++s) {
      const int c = s * 16 + j;
      const float pe = p0 * wp2c0[s] + p1 * wp2c1[s] + wp2b[s];
      out[row * 128 + c] = (ws[F3_OFF + gi * 128 + c] + pe) * wv;
    }
  }
}

// ===========================================================================
extern "C" void kernel_launch(void* const* d_in, const int* in_sizes, int n_in,
                              void* d_out, int out_size, void* d_ws,
                              size_t ws_size, hipStream_t stream) {
  (void)in_sizes; (void)n_in; (void)out_size; (void)ws_size;
  const float* points   = (const float*)d_in[0];
  const float* features = (const float*)d_in[1];
  const float* W1  = (const float*)d_in[2];
  const float* b1  = (const float*)d_in[3];
  const float* Wb  = (const float*)d_in[4];
  const float* bb  = (const float*)d_in[5];
  const float* Wp1 = (const float*)d_in[6];
  const float* gp  = (const float*)d_in[7];
  const float* bp  = (const float*)d_in[8];
  const float* Wp2 = (const float*)d_in[9];
  const float* bp2 = (const float*)d_in[10];
  const float* W2a = (const float*)d_in[11];
  const float* g2a = (const float*)d_in[12];
  const float* b2a = (const float*)d_in[13];
  const float* W2b = (const float*)d_in[14];
  const float* g2b = (const float*)d_in[15];
  const float* b2b = (const float*)d_in[16];
  const float* W2c = (const float*)d_in[17];
  const float* b2c = (const float*)d_in[18];
  const float* W3  = (const float*)d_in[19];
  const float* b3  = (const float*)d_in[20];
  float* out = (float*)d_out;
  float* ws  = (float*)d_ws;   // needs ~60 MB (see ws layout above)

  k_f3<<<256, 256, 0, stream>>>(features, W3, b3, ws);
  k1_energy<<<4096, 256, 0, stream>>>(points, features, W1, b1, Wb, bb, Wp1, out, ws);
  k_bn_fin<<<1, 32, 0, stream>>>(gp, bp, ws, PART1_OFF, 4096, BN1_SC, 2);
  k3_bn2a_stats<<<4096, 256, 0, stream>>>(Wp1, Wp2, bp2, W2a, out, ws);
  k_bn_fin<<<1, 32, 0, stream>>>(g2a, b2a, ws, PART2A_OFF, 4096, BN2A_SC, 32);
  k5_h2<<<4096, 256, 0, stream>>>(Wp1, Wp2, bp2, W2a, W2b, out, ws);
  k_bn_fin<<<1, 32, 0, stream>>>(g2b, b2b, ws, PART2B_OFF, 4096, BN2B_SC, 4);
  k7_out<<<2048, 256, 0, stream>>>(Wp1, Wp2, bp2, W2c, b2c, out, ws);
}
